// Generator_40604620816655
// MI455X (gfx1250) — matrix operand order repaired and checked by reading, not compiled, not run on hardware
//
#include <hip/hip_runtime.h>

// ---------------------------------------------------------------------------
// Persistent-workgroup autoregressive LSTM trajectory generator for gfx1250.
// - 64 workgroups x 256 threads (8 wave32). WG b owns batch rows [16b,16b+16).
// - All matmuls via v_wmma_f32_16x16x32_bf16 (bf16 in, f32 acc).
// - Weights bf16-converted + K-padded once into d_ws (L2-resident, ~2.2MB).
// - An opaque (asm-laundered) integer offset is added to the weight base
//   pointers each timestep: blocks LICM from hoisting the loop-invariant
//   weight loads out of the 200-step loop (which spilled to scratch in R1),
//   while preserving pointer provenance so loads lower to global_load (the
//   pointer-laundering variant in R2 degraded them to flat_load, which
//   couples LOADcnt with DScnt per the ISA).
// - Activations staged in LDS bf16; recurrent c-state held in VGPRs.
// ---------------------------------------------------------------------------

#define HH   256
#define TT   200
#define K0P  288   // 263 (=3+256+4) padded to 9*32
#define BT   16    // batch rows per workgroup

typedef __attribute__((ext_vector_type(16))) __bf16 v16bf;
typedef __attribute__((ext_vector_type(8)))  float  v8f;
typedef unsigned int u32x4 __attribute__((ext_vector_type(4)));
typedef unsigned short u16;

__device__ __forceinline__ u16 f2bf(float f) {
  unsigned int u = __float_as_uint(f);
  return (u16)((u + 0x7FFFu + ((u >> 16) & 1u)) >> 16);   // RNE
}
__device__ __forceinline__ float sigm(float x)  { return 1.f / (1.f + __expf(-x)); }
__device__ __forceinline__ float lrelu(float x) { return x > 0.f ? x : 0.2f * x; }

union AF { v16bf v; u32x4 q[2]; };

// A fragment (16x32 bf16, M x K), row-major source with 'stride' halfs per row.
// Layout: lanes<16 hold K 0-15, lanes>=16 hold K 16-31 (matches load_b).
__device__ __forceinline__ v16bf load_a(const u16* base, int stride, int k0,
                                        int l16, int hi) {
  AF a;
  const u16* p = base + l16 * stride + k0 + hi * 16;
  a.q[0] = *(const u32x4*)(p);
  a.q[1] = *(const u32x4*)(p + 8);
  return a.v;
}
// B fragment (32x16 bf16, K x N) from W stored row-major [N][stride] (i.e. W^T
// K-major per output unit). lane<16: col=n0+lane, K k0..k0+15 (32B contig);
// lane>=16: col=n0+lane-16, K k0+16..k0+31.
__device__ __forceinline__ v16bf load_b(const u16* w, int stride, int n0, int k0,
                                        int l16, int hi) {
  const u16* p = w + (n0 + l16) * stride + k0 + hi * 16;
  return *(const v16bf*)p;
}
#define WMMA_BF16(a, b, c) \
  __builtin_amdgcn_wmma_f32_16x16x32_bf16(false, (a), false, (b), (short)0, (c), false, false)

// --------------------------- weight prep ----------------------------------
__global__ void cvt_pad(const float* __restrict__ src, u16* __restrict__ dst,
                        int dstRows, int srcRows, int sk, int dk) {
  int i = blockIdx.x * blockDim.x + threadIdx.x;
  if (i >= dstRows * dk) return;
  int r = i / dk, k = i - r * dk;
  float v = (r < srcRows && k < sk) ? src[r * sk + k] : 0.f;
  dst[i] = f2bf(v);
}

// --------------------------- main kernel ----------------------------------
__global__ __launch_bounds__(256, 1) void traj_kernel(
    const float* __restrict__ start, const float* __restrict__ end_,
    const float* __restrict__ z, const float* __restrict__ init_in,
    const float* __restrict__ ce_w1, const float* __restrict__ ce_b1,
    const float* __restrict__ ce_g1, const float* __restrict__ ce_be1,
    const float* __restrict__ ce_w2, const float* __restrict__ ce_b2,
    const float* __restrict__ ce_g2, const float* __restrict__ ce_be2,
    const float* __restrict__ b_ih0, const float* __restrict__ b_hh0,
    const float* __restrict__ b_ih1, const float* __restrict__ b_hh1,
    const float* __restrict__ ob1,  const float* __restrict__ ob2,
    const u16* __restrict__ wih0_, const u16* __restrict__ whh0_,
    const u16* __restrict__ wih1_, const u16* __restrict__ whh1_,
    const u16* __restrict__ ow1_,  const u16* __restrict__ ow2_,
    float* __restrict__ out) {
  struct LoopS { u16 o1b[BT * 128]; float otile[BT * 16]; };
  struct InitS { float cib[BT * 72]; float tmp[BT * HH]; float cn1[BT * HH]; };
  __shared__ __align__(32) union { LoopS L; InitS I; } ovl;  // overlay scratch
  __shared__ __align__(32) u16 xb[BT * K0P];   // layer0 input, bf16
  __shared__ __align__(32) u16 h0b[BT * HH];   // h0, bf16
  __shared__ __align__(32) u16 h1b[BT * HH];   // h1, bf16
  __shared__ float posend[BT * 4];             // posx,posy,endx,endy
  __shared__ float lnst[BT * 2];               // LN mean / invstd

  const int tid  = threadIdx.x;
  const int wv   = tid >> 5;
  const int lane = tid & 31;
  const int l16  = lane & 15;
  const int hi   = lane >> 4;
  const int u0   = 32 * wv;           // hidden-unit base owned by this wave
  const int bt0  = blockIdx.x * BT;   // first batch row of this WG

  // ---- zero persistent LDS state ----
  for (int i = tid; i < BT * K0P; i += 256) xb[i] = 0;
  for (int i = tid; i < BT * HH; i += 256) { h0b[i] = 0; h1b[i] = 0; }
  __syncthreads();

  // ---- per-sample head + condition-encoder input ----
  if (tid < BT) {
    int m = tid, gs = bt0 + m;
    float sx = start[2 * gs], sy = start[2 * gs + 1];
    float ex = end_[2 * gs],  ey = end_[2 * gs + 1];
    float dx = ex - sx, dy = ey - sy;
    float dist = sqrtf(dx * dx + dy * dy + 1e-8f);
    float ang  = atan2f(dy, dx);
    float* cr = &ovl.I.cib[m * 72];
    cr[0] = sx; cr[1] = sy; cr[2] = ex; cr[3] = ey; cr[4] = dist; cr[5] = ang;
    for (int k = 0; k < 64; ++k) cr[6 + k] = z[gs * 64 + k];
    posend[m * 4 + 0] = sx; posend[m * 4 + 1] = sy;
    posend[m * 4 + 2] = ex; posend[m * 4 + 3] = ey;
    u16* xh = &xb[m * K0P];
    xh[0] = f2bf(init_in[0]); xh[1] = f2bf(init_in[1]); xh[2] = f2bf(init_in[2]);
    xh[259] = f2bf(dx); xh[260] = f2bf(dy);
    xh[261] = f2bf(dist); xh[262] = f2bf(ang);
  }
  __syncthreads();

  // ---- condition encoder (one-time, VALU): Lin(70->256)+LN+LReLU, Lin+LN+LReLU
  {
    const int s  = tid >> 4;           // sample
    const int ub = (tid & 15) * 16;    // 16 output units per thread
    for (int j = 0; j < 16; ++j) {
      int u = ub + j;
      const float* wrow = &ce_w1[u * 70];
      const float* cr = &ovl.I.cib[s * 72];
      float acc = ce_b1[u];
      for (int k = 0; k < 70; ++k) acc += cr[k] * wrow[k];
      ovl.I.tmp[s * HH + u] = acc;
    }
    __syncthreads();
    if (tid < BT) {
      float m0 = 0.f, m1 = 0.f;
      for (int k = 0; k < HH; ++k) { float v = ovl.I.tmp[tid * HH + k]; m0 += v; m1 += v * v; }
      m0 *= (1.f / HH); m1 = m1 * (1.f / HH) - m0 * m0;
      lnst[tid * 2] = m0; lnst[tid * 2 + 1] = rsqrtf(m1 + 1e-5f);
    }
    __syncthreads();
    for (int j = 0; j < 16; ++j) {
      int u = ub + j;
      float v = (ovl.I.tmp[s * HH + u] - lnst[s * 2]) * lnst[s * 2 + 1] * ce_g1[u] + ce_be1[u];
      ovl.I.cn1[s * HH + u] = lrelu(v);
    }
    __syncthreads();
    for (int j = 0; j < 16; ++j) {
      int u = ub + j;
      const float* wrow = &ce_w2[u * HH];
      const float* c1p = &ovl.I.cn1[s * HH];
      float acc = ce_b2[u];
      for (int k = 0; k < HH; ++k) acc += c1p[k] * wrow[k];
      ovl.I.tmp[s * HH + u] = acc;
    }
    __syncthreads();
    if (tid < BT) {
      float m0 = 0.f, m1 = 0.f;
      for (int k = 0; k < HH; ++k) { float v = ovl.I.tmp[tid * HH + k]; m0 += v; m1 += v * v; }
      m0 *= (1.f / HH); m1 = m1 * (1.f / HH) - m0 * m0;
      lnst[tid * 2] = m0; lnst[tid * 2 + 1] = rsqrtf(m1 + 1e-5f);
    }
    __syncthreads();
    for (int j = 0; j < 16; ++j) {
      int u = ub + j;
      float v = (ovl.I.tmp[s * HH + u] - lnst[s * 2]) * lnst[s * 2 + 1] * ce_g2[u] + ce_be2[u];
      xb[s * K0P + 3 + u] = f2bf(lrelu(v));   // cond occupies x[3..258]
    }
    __syncthreads();
  }

  // ---- per-lane biases held in registers (n depends only on l16) ----
  float bg0r[8], bg1r[8];
#pragma unroll
  for (int g = 0; g < 4; ++g)
#pragma unroll
    for (int b = 0; b < 2; ++b) {
      int idx = 256 * g + u0 + 16 * b + l16;
      bg0r[g * 2 + b] = b_ih0[idx] + b_hh0[idx];
      bg1r[g * 2 + b] = b_ih1[idx] + b_hh1[idx];
    }
  float bo1r = ob1[16 * wv + l16];
  float bo2r = (l16 < 3) ? ob2[l16] : 0.f;

  // cell state in registers: [subtile b][row r], m = r + 8*hi, n = l16
  float c0r[2][8], c1r[2][8];
#pragma unroll
  for (int b = 0; b < 2; ++b)
#pragma unroll
    for (int r = 0; r < 8; ++r) { c0r[b][r] = 0.f; c1r[b][r] = 0.f; }

  // ================= recurrent loop =================
  for (int t = 0; t < TT; ++t) {
    // Opaque zero offset: loads below depend on a value redefined every
    // iteration, so LICM cannot hoist them; provenance of the kernarg
    // pointers is preserved -> global_load, LOADcnt only.
    size_t zofs = 0;
    asm volatile("" : "+s"(zofs));
    const u16* wih0 = wih0_ + zofs;
    const u16* whh0 = whh0_ + zofs;
    const u16* wih1 = wih1_ + zofs;
    const u16* whh1 = whh1_ + zofs;
    const u16* ow1  = ow1_  + zofs;
    const u16* ow2  = ow2_  + zofs;

    // ---- layer0 gates: x(288) @ wih0^T + h0(256) @ whh0^T ----
    v8f acc[8];
#pragma unroll
    for (int i = 0; i < 8; ++i) {
      v8f a;
#pragma unroll
      for (int r = 0; r < 8; ++r) a[r] = bg0r[i];
      acc[i] = a;
    }
#pragma unroll 2
    for (int kt = 0; kt < 9; ++kt) {
      int k0 = kt * 32;
      v16bf bf[8];
#pragma unroll
      for (int g = 0; g < 4; ++g)
#pragma unroll
        for (int b = 0; b < 2; ++b)
          bf[g * 2 + b] = load_b(wih0, K0P, 256 * g + u0 + 16 * b, k0, l16, hi);
      v16bf af = load_a(xb, K0P, k0, l16, hi);
#pragma unroll
      for (int i = 0; i < 8; ++i) acc[i] = WMMA_BF16(af, bf[i], acc[i]);
    }
#pragma unroll 2
    for (int kt = 0; kt < 8; ++kt) {
      int k0 = kt * 32;
      v16bf bf[8];
#pragma unroll
      for (int g = 0; g < 4; ++g)
#pragma unroll
        for (int b = 0; b < 2; ++b)
          bf[g * 2 + b] = load_b(whh0, HH, 256 * g + u0 + 16 * b, k0, l16, hi);
      v16bf af = load_a(h0b, HH, k0, l16, hi);
#pragma unroll
      for (int i = 0; i < 8; ++i) acc[i] = WMMA_BF16(af, bf[i], acc[i]);
    }
    __syncthreads();   // all waves done reading old h0b
    // elementwise LSTM cell 0 + publish new h0
#pragma unroll
    for (int b = 0; b < 2; ++b)
#pragma unroll
      for (int r = 0; r < 8; ++r) {
        float iv = sigm(acc[0 * 2 + b][r]);
        float fv = sigm(acc[1 * 2 + b][r]);
        float gv = tanhf(acc[2 * 2 + b][r]);
        float ov = sigm(acc[3 * 2 + b][r]);
        float c = fv * c0r[b][r] + iv * gv;
        c0r[b][r] = c;
        float h = ov * tanhf(c);
        h0b[(r + 8 * hi) * HH + u0 + 16 * b + l16] = f2bf(h);
      }
    __syncthreads();

    // ---- layer1 gates: h0_new @ wih1^T + h1_old @ whh1^T ----
    v8f ac2[8];
#pragma unroll
    for (int i = 0; i < 8; ++i) {
      v8f a;
#pragma unroll
      for (int r = 0; r < 8; ++r) a[r] = bg1r[i];
      ac2[i] = a;
    }
#pragma unroll 2
    for (int kt = 0; kt < 8; ++kt) {
      int k0 = kt * 32;
      v16bf bf[8];
#pragma unroll
      for (int g = 0; g < 4; ++g)
#pragma unroll
        for (int b = 0; b < 2; ++b)
          bf[g * 2 + b] = load_b(wih1, HH, 256 * g + u0 + 16 * b, k0, l16, hi);
      v16bf af = load_a(h0b, HH, k0, l16, hi);
#pragma unroll
      for (int i = 0; i < 8; ++i) ac2[i] = WMMA_BF16(af, bf[i], ac2[i]);
    }
#pragma unroll 2
    for (int kt = 0; kt < 8; ++kt) {
      int k0 = kt * 32;
      v16bf bf[8];
#pragma unroll
      for (int g = 0; g < 4; ++g)
#pragma unroll
        for (int b = 0; b < 2; ++b)
          bf[g * 2 + b] = load_b(whh1, HH, 256 * g + u0 + 16 * b, k0, l16, hi);
      v16bf af = load_a(h1b, HH, k0, l16, hi);
#pragma unroll
      for (int i = 0; i < 8; ++i) ac2[i] = WMMA_BF16(af, bf[i], ac2[i]);
    }
    __syncthreads();   // all waves done reading old h1b
#pragma unroll
    for (int b = 0; b < 2; ++b)
#pragma unroll
      for (int r = 0; r < 8; ++r) {
        float iv = sigm(ac2[0 * 2 + b][r]);
        float fv = sigm(ac2[1 * 2 + b][r]);
        float gv = tanhf(ac2[2 * 2 + b][r]);
        float ov = sigm(ac2[3 * 2 + b][r]);
        float c = fv * c1r[b][r] + iv * gv;
        c1r[b][r] = c;
        float h = ov * tanhf(c);
        h1b[(r + 8 * hi) * HH + u0 + 16 * b + l16] = f2bf(h);
      }
    __syncthreads();

    // ---- MLP hidden: lrelu(h1 @ ow1^T + ob1); wave w owns units [16w,16w+16)
    {
      v8f am;
#pragma unroll
      for (int r = 0; r < 8; ++r) am[r] = bo1r;
#pragma unroll 2
      for (int kt = 0; kt < 8; ++kt) {
        int k0 = kt * 32;
        v16bf bf = load_b(ow1, HH, 16 * wv, k0, l16, hi);
        v16bf af = load_a(h1b, HH, k0, l16, hi);
        am = WMMA_BF16(af, bf, am);
      }
#pragma unroll
      for (int r = 0; r < 8; ++r)
        ovl.L.o1b[(r + 8 * hi) * 128 + 16 * wv + l16] = f2bf(lrelu(am[r]));
    }
    __syncthreads();

    // ---- output head + state update (wave 0 only) ----
    if (wv == 0) {
      v8f ao;
#pragma unroll
      for (int r = 0; r < 8; ++r) ao[r] = bo2r;
#pragma unroll
      for (int kt = 0; kt < 4; ++kt) {
        int k0 = kt * 32;
        v16bf bf = load_b(ow2, 128, 0, k0, l16, hi);
        v16bf af = load_a(ovl.L.o1b, 128, k0, l16, hi);
        ao = WMMA_BF16(af, bf, ao);
      }
#pragma unroll
      for (int r = 0; r < 8; ++r)
        ovl.L.otile[(r + 8 * hi) * 16 + l16] = ao[r];
      // same-wave DS ops are ordered: safe to read back immediately
      if (lane < 16) {
        int m = lane, gs = bt0 + m;
        float o0 = ovl.L.otile[m * 16 + 0];
        float o1 = ovl.L.otile[m * 16 + 1];
        float o2 = ovl.L.otile[m * 16 + 2];
        float o2s = (o2 > 20.f ? o2 : log1pf(__expf(o2))) + 0.001f;  // softplus+eps
        float px = posend[m * 4 + 0] + o0;
        float py = posend[m * 4 + 1] + o1;
        posend[m * 4 + 0] = px; posend[m * 4 + 1] = py;
        float rx = posend[m * 4 + 2] - px;
        float ry = posend[m * 4 + 3] - py;
        float rd = sqrtf(rx * rx + ry * ry + 1e-8f);
        float ra = atan2f(ry, rx);
        float* op = out + ((size_t)gs * TT + t) * 3;
        op[0] = o0; op[1] = o1; op[2] = o2s;
        u16* xh = &xb[m * K0P];
        xh[0] = f2bf(o0); xh[1] = f2bf(o1); xh[2] = f2bf(o2s);
        xh[259] = f2bf(rx); xh[260] = f2bf(ry);
        xh[261] = f2bf(rd); xh[262] = f2bf(ra);
      }
    }
    __syncthreads();   // publish new x head before next step
  }
}

// --------------------------- launcher -------------------------------------
extern "C" void kernel_launch(void* const* d_in, const int* in_sizes, int n_in,
                              void* d_out, int out_size, void* d_ws, size_t ws_size,
                              hipStream_t stream) {
  (void)in_sizes; (void)n_in; (void)out_size; (void)ws_size;
  // ws layout (bf16 halfs)
  u16* ws    = (u16*)d_ws;
  u16* wih0b = ws;                         // 1024 x 288
  u16* whh0b = wih0b + 1024 * 288;         // 1024 x 256
  u16* wih1b = whh0b + 1024 * 256;         // 1024 x 256
  u16* whh1b = wih1b + 1024 * 256;         // 1024 x 256
  u16* ow1b  = whh1b + 1024 * 256;         // 128  x 256
  u16* ow2b  = ow1b  + 128 * 256;          // 16   x 128 (rows 3..15 zero)

  auto cvt = [&](const void* s, u16* dst, int dr, int sr, int sk, int dk) {
    int n = dr * dk;
    cvt_pad<<<(n + 255) / 256, 256, 0, stream>>>((const float*)s, dst, dr, sr, sk, dk);
  };
  cvt(d_in[12], wih0b, 1024, 1024, 263, 288);   // w_ih0
  cvt(d_in[13], whh0b, 1024, 1024, 256, 256);   // w_hh0
  cvt(d_in[16], wih1b, 1024, 1024, 256, 256);   // w_ih1
  cvt(d_in[17], whh1b, 1024, 1024, 256, 256);   // w_hh1
  cvt(d_in[20], ow1b,  128,  128,  256, 256);   // ow1
  cvt(d_in[22], ow2b,  16,   3,    128, 128);   // ow2 (zero-pad rows)

  traj_kernel<<<64, 256, 0, stream>>>(
      (const float*)d_in[0],  (const float*)d_in[1],  (const float*)d_in[2],
      (const float*)d_in[3],
      (const float*)d_in[4],  (const float*)d_in[5],  (const float*)d_in[6],
      (const float*)d_in[7],  (const float*)d_in[8],  (const float*)d_in[9],
      (const float*)d_in[10], (const float*)d_in[11],
      (const float*)d_in[14], (const float*)d_in[15],
      (const float*)d_in[18], (const float*)d_in[19],
      (const float*)d_in[21], (const float*)d_in[23],
      wih0b, whh0b, wih1b, whh1b, ow1b, ow2b,
      (float*)d_out);
}
